// ReformerLayer_51153060496048
// MI455X (gfx1250) — matrix operand order repaired and checked by reading, not compiled, not run on hardware
//
#include <hip/hip_runtime.h>
#include <hip/hip_bf16.h>

// ---------------------------------------------------------------------------
// Reformer LSH attention layer for MI455X (gfx1250, wave32, WMMA).
// B=8, SEQ=2048, D=512, H=8, d=64, bucket=4, n_hashes=4, n_buckets=512.
//   0) conv_bf16 / conv_bf16_t : pre-convert X (row-major) and W^T to bf16
//   1) gemm_bf16 (WMMA, async double-buffered LDS): qk = Q@Wqk, v = Q@Wv
//   2) lsh_hash  : rotated = qk @ rotations, argmax over [r,-r] -> sort keys
//   3) lsh_sort  : per head-batch bitonic sort of 8192 unique 32-bit keys
//   4) lsh_attend(WMMA): 4-chunk-batched dots/softmax/probs@V
//   5) combine   : unsort + logsumexp merge of 4 hash rounds (emit bf16)
//   6) gemm_bf16 : out = O@Wout + b_out
// ---------------------------------------------------------------------------

typedef __attribute__((ext_vector_type(16))) __bf16 v16bf;
typedef __attribute__((ext_vector_type(8)))  __bf16 v8bf;
typedef __attribute__((ext_vector_type(4)))  __bf16 v4bf;
typedef __attribute__((ext_vector_type(8)))  float  v8f;
typedef __attribute__((ext_vector_type(4)))  int    v4i;

#define SEQ_      2048
#define DHEAD     64
#define NHASH     4
#define NBKT      512
#define NSORT     8192      // NHASH * SEQ
#define NCHUNK    2048      // NHASH * NBKT
#define NBM       64        // B * H

// CDNA5 async global->LDS path (ASYNCcnt), with synchronous fallback.
#if defined(__gfx1250__) && __has_builtin(__builtin_amdgcn_global_load_async_to_lds_b128) && __has_builtin(__builtin_amdgcn_s_wait_asynccnt)
#define USE_ASYNC_LDS 1
#else
#define USE_ASYNC_LDS 0
#endif

static __device__ __forceinline__ __bf16 f2bf(float f) {
  unsigned u = __builtin_bit_cast(unsigned, f);
  u += 0x7FFFu + ((u >> 16) & 1u);          // round-to-nearest-even
  unsigned short h = (unsigned short)(u >> 16);
  return __builtin_bit_cast(__bf16, h);
}

static __device__ __forceinline__ v8f wmma_bf16(v16bf a, v16bf b, v8f c) {
  return __builtin_amdgcn_wmma_f32_16x16x32_bf16(
      false, a, false, b, (short)0, c, false, false);
}

union U16 { v16bf v; v8bf h[2]; };

// one 16-byte global->LDS transfer
static __device__ __forceinline__ void copy16(const __bf16* g, __bf16* l) {
#if USE_ASYNC_LDS
  __builtin_amdgcn_global_load_async_to_lds_b128(
      (__attribute__((address_space(1))) v4i*)g,
      (__attribute__((address_space(3))) v4i*)l, 0, 0);
#else
  *(v8bf*)l = *(const v8bf*)g;
#endif
}
static __device__ __forceinline__ void wait_async() {
#if USE_ASYNC_LDS
  __builtin_amdgcn_s_wait_asynccnt(0);
#endif
}

// ---------------------------------------------------------------------------
// Kernel 0a: f32 -> bf16 straight convert (4 elements / thread)
// ---------------------------------------------------------------------------
__global__ __launch_bounds__(256)
void conv_bf16(const float* __restrict__ in, __bf16* __restrict__ out) {
  int i = blockIdx.x * 256 + threadIdx.x;
  float4 vld = ((const float4*)in)[i];
  v4bf o;
  o[0] = f2bf(vld.x); o[1] = f2bf(vld.y); o[2] = f2bf(vld.z); o[3] = f2bf(vld.w);
  ((v4bf*)out)[i] = o;
}

// ---------------------------------------------------------------------------
// Kernel 0b: f32 (KxN) -> bf16 transposed (NxK), K=N=512
// ---------------------------------------------------------------------------
__global__ __launch_bounds__(256)
void conv_bf16_t(const float* __restrict__ in, __bf16* __restrict__ out) {
  int idx = blockIdx.x * 256 + threadIdx.x;     // 262144 threads
  int n = idx >> 9, k = idx & 511;
  out[(size_t)n * 512 + k] = f2bf(in[(size_t)k * 512 + n]);
}

// ---------------------------------------------------------------------------
// Kernel 1/6: C = A(MxK bf16) @ Bt^T (Bt is NxK bf16) [+ bias], WMMA f32 acc.
// Async double-buffered LDS tiles. Block: 256 thr = 8 waves; tile 128x64.
// head_split!=0 writes Y in (B*H, SEQ, DHEAD) layout, else flat (M,N)+bias.
// ---------------------------------------------------------------------------
__global__ __launch_bounds__(256)
void gemm_bf16(const __bf16* __restrict__ A, const __bf16* __restrict__ Bt,
               const float* __restrict__ bias, float* __restrict__ Y,
               int M, int K, int N, int head_split) {
  __shared__ __attribute__((aligned(16))) __bf16 As[2][128][40];  // 128x32 used
  __shared__ __attribute__((aligned(16))) __bf16 Bs[2][64][40];   // 64x32 used

  const int tid  = threadIdx.x;
  const int wave = tid >> 5;
  const int lane = tid & 31;
  const int g    = lane >> 4;
  const int ln   = lane & 15;
  const int wm   = wave & 3;       // 32-row group
  const int wn   = wave >> 2;      // 32-col group
  const int rowBase = blockIdx.x * 128;
  const int colBase = blockIdx.y * 64;

  v8f acc[2][2];
  for (int ti = 0; ti < 2; ++ti)
    for (int tj = 0; tj < 2; ++tj)
      for (int r = 0; r < 8; ++r) acc[ti][tj][r] = 0.0f;

  // stage K-tile `kk` into buffer `buf` (pure 16B copies)
  auto stage = [&](int kk, int buf) {
    // A tile: 128 rows x 32 halves = 512 chunks of 8 halves
    for (int c = tid; c < 512; c += 256) {
      int r = c >> 2, s = c & 3;
      copy16(&A[(size_t)(rowBase + r) * K + kk + s * 8], &As[buf][r][s * 8]);
    }
    // B tile: 64 cols x 32 halves = 256 chunks
    {
      int n = tid >> 2, s = tid & 3;
      copy16(&Bt[(size_t)(colBase + n) * K + kk + s * 8], &Bs[buf][n][s * 8]);
    }
  };

  const int nsteps = K >> 5;
  stage(0, 0);
  for (int step = 0; step < nsteps; ++step) {
    const int buf = step & 1;
    wait_async();
    __syncthreads();
    if (step + 1 < nsteps) {
      stage((step + 1) << 5, buf ^ 1);
      __builtin_prefetch(&A[(size_t)(rowBase + (tid >> 1)) * K + ((step + 1) << 5)], 0, 1);
    }
    for (int ti = 0; ti < 2; ++ti) {
      const int mrow = wm * 32 + ti * 16 + ln;
      U16 a;   // lane layout: halves [8g,8g+8) then [16+8g,16+8g+8)
      a.h[0] = *(const v8bf*)&As[buf][mrow][8 * g];
      a.h[1] = *(const v8bf*)&As[buf][mrow][16 + 8 * g];
      for (int tj = 0; tj < 2; ++tj) {
        const int ncol = wn * 32 + tj * 16 + ln;
        U16 b;   // lane layout: halves [8g,8g+8) then [16+8g,16+8g+8) of column ncol
        b.h[0] = *(const v8bf*)&Bs[buf][ncol][8 * g];
        b.h[1] = *(const v8bf*)&Bs[buf][ncol][16 + 8 * g];
        acc[ti][tj] = wmma_bf16(a.v, b.v, acc[ti][tj]);
      }
    }
  }

  // store: C/D layout -> lane holds (row r + 8g, col ln)
  for (int ti = 0; ti < 2; ++ti)
    for (int tj = 0; tj < 2; ++tj)
#pragma unroll
      for (int r = 0; r < 8; ++r) {
        int grow = rowBase + wm * 32 + ti * 16 + r + 8 * g;
        int gcol = colBase + wn * 32 + tj * 16 + ln;
        float val = acc[ti][tj][r];
        if (head_split) {
          int b = grow >> 11, nloc = grow & 2047;
          int head = gcol >> 6, dc = gcol & 63;
          Y[(((size_t)b * 8 + head) * SEQ_ + nloc) * DHEAD + dc] = val;
        } else {
          Y[(size_t)grow * N + gcol] = val + bias[gcol];
        }
      }
}

// ---------------------------------------------------------------------------
// Kernel 2/6: LSH hashing. bucket = argmax over [rotated, -rotated];
// key = (bucket + h*512)*2048 + t.  grid (8 tok-blocks, 4 hashes, 64 bm).
// ---------------------------------------------------------------------------
__global__ __launch_bounds__(256)
void lsh_hash(const float* __restrict__ qk, const float* __restrict__ rot,
              unsigned* __restrict__ keys) {
  __shared__ float Rs[64][128];    // 32 KB: half the rotation columns
  const int tid = threadIdx.x;
  const int bm  = blockIdx.z;
  const int h   = blockIdx.y;
  const int tok = blockIdx.x * 256 + tid;

  float q[64];
  const float4* qrow4 = (const float4*)(qk + ((size_t)bm * SEQ_ + tok) * DHEAD);
#pragma unroll
  for (int f = 0; f < 16; ++f) *(float4*)&q[f * 4] = qrow4[f];

  float best = -3.0e38f;
  int bidx = 0;
  for (int pass = 0; pass < 2; ++pass) {
    __syncthreads();
    for (int idx = tid; idx < 64 * 128; idx += 256) {
      int f = idx >> 7, i = idx & 127;
      Rs[f][i] = rot[(size_t)f * (NHASH * 256) + h * 256 + pass * 128 + i];
    }
    __syncthreads();
    for (int i = 0; i < 128; ++i) {
      float s = 0.0f;
#pragma unroll
      for (int f = 0; f < 64; ++f) s += q[f] * Rs[f][i];
      int ip = pass * 128 + i;
      if (s > best)  { best = s;  bidx = ip; }
      if (-s > best) { best = -s; bidx = 256 + ip; }
    }
  }
  unsigned key = ((unsigned)(bidx + h * NBKT) << 11) | (unsigned)tok;
  keys[(size_t)bm * NSORT + h * SEQ_ + tok] = key;
}

// ---------------------------------------------------------------------------
// Kernel 3/6: bitonic sort of 8192 unique keys per bm in LDS.
// ---------------------------------------------------------------------------
__global__ __launch_bounds__(256)
void lsh_sort(const unsigned* __restrict__ keys, int* __restrict__ st,
              int* __restrict__ undo) {
  __shared__ unsigned sk[NSORT];   // 32 KB
  const int tid = threadIdx.x;
  const int bm  = blockIdx.x;

  for (int i = tid; i < NSORT; i += 256) sk[i] = keys[(size_t)bm * NSORT + i];
  __syncthreads();

  for (int k = 2; k <= NSORT; k <<= 1) {
    for (int j = k >> 1; j > 0; j >>= 1) {
      for (int i = tid; i < NSORT; i += 256) {
        int l = i ^ j;
        if (l > i) {
          unsigned a = sk[i], b = sk[l];
          bool asc = ((i & k) == 0);
          if ((asc && a > b) || (!asc && a < b)) { sk[i] = b; sk[l] = a; }
        }
      }
      __syncthreads();
    }
  }

  for (int i = tid; i < NSORT; i += 256) {
    unsigned key = sk[i];
    int pos    = (int)(key & 2047u);
    int bucket = (int)(key >> 11);
    int h      = bucket >> 9;
    st[(size_t)bm * NSORT + i] = pos;
    undo[(size_t)bm * NSORT + h * SEQ_ + pos] = i;
  }
}

// ---------------------------------------------------------------------------
// Kernel 4/6: chunked attention. One wave per group of 4 chunks
// (16 queries, 32 kv = self+prev per chunk). WMMA dots and probs@V.
// ---------------------------------------------------------------------------
__global__ __launch_bounds__(256)
void lsh_attend(const float* __restrict__ qk, const float* __restrict__ vv,
                const int* __restrict__ st, float* __restrict__ so,
                float* __restrict__ slog) {
  __shared__ float  dots_sh[8][16][32];
  __shared__ __attribute__((aligned(16))) __bf16 probs_sh[8][16][32];
  __shared__ int    stq_sh[8][16];
  __shared__ int    stkv_sh[8][32];
  __shared__ float  rinv_sh[8][32];

  const int tid  = threadIdx.x;
  const int w    = tid >> 5;
  const int lane = tid & 31;
  const int g    = lane >> 4;
  const int ln   = lane & 15;
  const int bm   = blockIdx.y;
  const int group = blockIdx.x * 8 + w;
  const int qbase = group * 16;
  const int cbase = group * 4;

  if (g == 0) stq_sh[w][ln] = st[(size_t)bm * NSORT + qbase + ln];
  {
    int kvi = lane;
    int cj = kvi >> 3, o = kvi & 7;
    int sc = (o < 4) ? (cbase + cj) : ((cbase + cj + NCHUNK - 1) & (NCHUNK - 1));
    int srow = sc * 4 + (o & 3);
    int pos = st[(size_t)bm * NSORT + srow];
    stkv_sh[w][kvi] = pos;
    const float4* kr = (const float4*)(qk + ((size_t)bm * SEQ_ + pos) * DHEAD);
    float s = 0.0f;
#pragma unroll
    for (int f = 0; f < 16; ++f) {
      float4 t = kr[f];
      s += t.x * t.x + t.y * t.y + t.z * t.z + t.w * t.w;
    }
    float nrm = fmaxf(sqrtf(s), 1e-12f);
    rinv_sh[w][kvi] = 1.0f / nrm;
  }
  __syncthreads();

  // A fragments: bq 16x64 (two K=32 halves); each lane needs contiguous
  // f32 runs [8g,8g+8), [16+8g,+8), [32+8g,+8), [48+8g,+8) -> float4 loads.
  const float4* q4 = (const float4*)(qk + ((size_t)bm * SEQ_ + stq_sh[w][ln]) * DHEAD);
  v16bf a0, a1;
  {
    float fa[16], fb[16];
    *(float4*)&fa[0]  = q4[2 * g];     *(float4*)&fa[4]  = q4[2 * g + 1];
    *(float4*)&fa[8]  = q4[4 + 2 * g]; *(float4*)&fa[12] = q4[5 + 2 * g];
    *(float4*)&fb[0]  = q4[8 + 2 * g]; *(float4*)&fb[4]  = q4[9 + 2 * g];
    *(float4*)&fb[8]  = q4[12 + 2 * g];*(float4*)&fb[12] = q4[13 + 2 * g];
#pragma unroll
    for (int i = 0; i < 16; ++i) { a0[i] = f2bf(fa[i]); a1[i] = f2bf(fb[i]); }
  }

  // dots = bq @ bk^T * d^-0.5 : 2 N-tiles x 2 K-steps of WMMA
  v8f d[2];
  for (int tt = 0; tt < 2; ++tt) {
    for (int r = 0; r < 8; ++r) d[tt][r] = 0.0f;
    int kv = tt * 16 + ln;
    const float4* k4 = (const float4*)(qk + ((size_t)bm * SEQ_ + stkv_sh[w][kv]) * DHEAD);
    float ri = rinv_sh[w][kv];
    v16bf b0, b1;
    {
      float kb0[16], kb1[16];   // runs [8g,+8), [16+8g,+8), [32+8g,+8), [48+8g,+8)
      *(float4*)&kb0[0]  = k4[2 * g];     *(float4*)&kb0[4]  = k4[2 * g + 1];
      *(float4*)&kb0[8]  = k4[4 + 2 * g]; *(float4*)&kb0[12] = k4[5 + 2 * g];
      *(float4*)&kb1[0]  = k4[8 + 2 * g];     *(float4*)&kb1[4]  = k4[9 + 2 * g];
      *(float4*)&kb1[8]  = k4[12 + 2 * g]; *(float4*)&kb1[12] = k4[13 + 2 * g];
#pragma unroll
      for (int i = 0; i < 16; ++i) {
        b0[i] = f2bf(kb0[i] * ri);
        b1[i] = f2bf(kb1[i] * ri);
      }
    }
    d[tt] = wmma_bf16(a0, b0, d[tt]);
    d[tt] = wmma_bf16(a1, b1, d[tt]);
  }

  // mask (cross-chunk -> -1e30, self -> -5e4) and park in LDS
  for (int tt = 0; tt < 2; ++tt)
#pragma unroll
    for (int r = 0; r < 8; ++r) {
      int m = r + 8 * g;
      int c = tt * 16 + ln;
      float val = d[tt][r] * 0.125f;   // 64^-0.5
      if ((c >> 3) != (m >> 2))      val = -1.0e30f;
      else if (stq_sh[w][m] == stkv_sh[w][c]) val = -5.0e4f;
      dots_sh[w][m][c] = val;
    }
  __syncthreads();

  // row softmax + logsumexp (one lane per row)
  if (g == 0) {
    int m = ln;
    float mx = -3.0e38f;
    for (int c = 0; c < 32; ++c) mx = fmaxf(mx, dots_sh[w][m][c]);
    float s = 0.0f;
    for (int c = 0; c < 32; ++c) s += __expf(dots_sh[w][m][c] - mx);
    float lse = mx + __logf(s);
    float inv = 1.0f / s;
    for (int c = 0; c < 32; ++c)
      probs_sh[w][m][c] = f2bf(__expf(dots_sh[w][m][c] - mx) * inv);
    slog[(size_t)bm * NSORT + qbase + m] = lse;
  }
  __syncthreads();

  // bo = probs(16x32) @ bv(32x64): A=probs (K=32), 4 N-tiles of WMMA
  U16 ap;
  ap.h[0] = *(const v8bf*)&probs_sh[w][ln][8 * g];
  ap.h[1] = *(const v8bf*)&probs_sh[w][ln][16 + 8 * g];
  for (int nt = 0; nt < 4; ++nt) {
    int col = nt * 16 + ln;
    v16bf bvf;   // element i: bv row (kv index) 8g+i+8*(i>>3), fixed column -> scattered
#pragma unroll
    for (int i = 0; i < 16; ++i)
      bvf[i] = f2bf(vv[((size_t)bm * SEQ_ + stkv_sh[w][8 * g + i + ((i >> 3) << 3)]) * DHEAD + col]);
    v8f o;
    for (int r = 0; r < 8; ++r) o[r] = 0.0f;
    o = wmma_bf16(ap.v, bvf, o);
#pragma unroll
    for (int r = 0; r < 8; ++r)
      so[((size_t)bm * NSORT + qbase + r + 8 * g) * DHEAD + col] = o[r];
  }
}

// ---------------------------------------------------------------------------
// Kernel 5/6: unsort + logsumexp-combine 4 hash rounds -> bf16 (B,SEQ,D)
// ---------------------------------------------------------------------------
__global__ __launch_bounds__(256)
void lsh_combine(const float* __restrict__ so, const float* __restrict__ slog,
                 const int* __restrict__ undo, __bf16* __restrict__ obf) {
  int gidx = blockIdx.x * 256 + threadIdx.x;     // 64*2048 total
  int bm = gidx >> 11, t = gidx & 2047;
  int b = bm >> 3, head = bm & 7;

  int jj[NHASH];
  float lg[NHASH];
  float mx = -3.0e38f;
#pragma unroll
  for (int h = 0; h < NHASH; ++h) {
    jj[h] = undo[(size_t)bm * NSORT + h * SEQ_ + t];
    lg[h] = slog[(size_t)bm * NSORT + jj[h]];
    mx = fmaxf(mx, lg[h]);
  }
  float s = 0.0f;
#pragma unroll
  for (int h = 0; h < NHASH; ++h) { lg[h] = __expf(lg[h] - mx); s += lg[h]; }
  float inv = 1.0f / s;

  __bf16* orow = obf + ((size_t)b * SEQ_ + t) * 512 + head * DHEAD;
  for (int dc = 0; dc < DHEAD; ++dc) {
    float acc = 0.0f;
#pragma unroll
    for (int h = 0; h < NHASH; ++h)
      acc += lg[h] * so[((size_t)bm * NSORT + jj[h]) * DHEAD + dc];
    orow[dc] = f2bf(acc * inv);
  }
}

// ---------------------------------------------------------------------------
extern "C" void kernel_launch(void* const* d_in, const int* in_sizes, int n_in,
                              void* d_out, int out_size, void* d_ws, size_t ws_size,
                              hipStream_t stream) {
  const float* q    = (const float*)d_in[0];
  const float* Wqk  = (const float*)d_in[3];
  const float* Wv   = (const float*)d_in[4];
  const float* Wout = (const float*)d_in[5];
  const float* bout = (const float*)d_in[6];
  const float* rot  = (const float*)d_in[7];
  float* out = (float*)d_out;

  char* ws = (char*)d_ws;
  size_t off = 0;
  auto carve = [&](size_t bytes) -> char* {
    char* p = ws + off;
    off += (bytes + 255) & ~(size_t)255;
    return p;
  };
  float*    qk    = (float*)   carve((size_t)NBM * SEQ_ * DHEAD * 4);   //  32 MB
  float*    vv    = (float*)   carve((size_t)NBM * SEQ_ * DHEAD * 4);   //  32 MB
  unsigned* keys  = (unsigned*)carve((size_t)NBM * NSORT * 4);          //   2 MB
  int*      st    = (int*)     carve((size_t)NBM * NSORT * 4);          //   2 MB
  int*      undo  = (int*)     carve((size_t)NBM * NSORT * 4);          //   2 MB
  float*    slog  = (float*)   carve((size_t)NBM * NSORT * 4);          //   2 MB
  float*    so    = (float*)   carve((size_t)NBM * NSORT * DHEAD * 4);  // 128 MB
  __bf16*   Xbf   = (__bf16*)  carve((size_t)8 * SEQ_ * 512 * 2);       //  16 MB
  __bf16*   WtQK  = (__bf16*)  carve((size_t)512 * 512 * 2);            // 0.5 MB
  __bf16*   WtV   = (__bf16*)  carve((size_t)512 * 512 * 2);            // 0.5 MB
  __bf16*   WtO   = (__bf16*)  carve((size_t)512 * 512 * 2);            // 0.5 MB
  __bf16*   Obf   = (__bf16*)  carve((size_t)8 * SEQ_ * 512 * 2);       //  16 MB

  const int M = 8 * SEQ_;   // 16384

  // 0) operand conversion (X row-major bf16, weights transposed bf16)
  conv_bf16<<<(M * 512 / 4) / 256, 256, 0, stream>>>(q, Xbf);
  conv_bf16_t<<<(512 * 512) / 256, 256, 0, stream>>>(Wqk, WtQK);
  conv_bf16_t<<<(512 * 512) / 256, 256, 0, stream>>>(Wv,  WtV);
  conv_bf16_t<<<(512 * 512) / 256, 256, 0, stream>>>(Wout, WtO);
  // 1) projections (head-split outputs)
  gemm_bf16<<<dim3(M / 128, 512 / 64), 256, 0, stream>>>(Xbf, WtQK, bout, qk, M, 512, 512, 1);
  gemm_bf16<<<dim3(M / 128, 512 / 64), 256, 0, stream>>>(Xbf, WtV,  bout, vv, M, 512, 512, 1);
  // 2) LSH hashing
  lsh_hash<<<dim3(SEQ_ / 256, NHASH, NBM), 256, 0, stream>>>(qk, rot, keys);
  // 3) sort per head-batch
  lsh_sort<<<NBM, 256, 0, stream>>>(keys, st, undo);
  // 4) chunked attention
  lsh_attend<<<dim3(NCHUNK / 4 / 8, NBM), 256, 0, stream>>>(qk, vv, st, so, slog);
  // 5) combine hash rounds -> bf16 activations
  lsh_combine<<<(NBM * SEQ_) / 256, 256, 0, stream>>>(so, slog, undo, Obf);
  // 6) output projection (+bias, flat layout)
  gemm_bf16<<<dim3(M / 128, 512 / 64), 256, 0, stream>>>(Obf, WtO, bout, out, M, 512, 512, 0);
}
